// SwinDiTBlock_23441931502165
// MI455X (gfx1250) — matrix operand order repaired and checked by reading, not compiled, not run on hardware
//
#include <hip/hip_runtime.h>

// ---------------------------------------------------------------------------
// Swin-DiT block for MI455X (gfx1250, wave32, WMMA 16x16x32 f16).
// ---------------------------------------------------------------------------

typedef __attribute__((ext_vector_type(4)))  _Float16 v4h;
typedef __attribute__((ext_vector_type(8)))  _Float16 v8h;
typedef __attribute__((ext_vector_type(16))) _Float16 v16h;
typedef __attribute__((ext_vector_type(8)))  float    v8f;

#define DIMC     512
#define NHEAD    8
#define HDIM     64
#define NWIN     256      // tokens per window
#define LTOK     65536    // total tokens (16*64*64)
#define HIDDEN   2048

#define WMMA_F16(A, B, C) \
  __builtin_amdgcn_wmma_f32_16x16x32_f16(false, (A), false, (B), (short)0, (C), false, false)

__device__ __forceinline__ v16h frag_cat(v8h lo, v8h hi) {
  return __builtin_shufflevector(lo, hi, 0,1,2,3,4,5,6,7,8,9,10,11,12,13,14,15);
}
// A-fragment (16x32 f16): lane holds row = lane&15, kb = (lane>>4)*8;
// data at [k0+kb, +8) and [k0+16+kb, +8). Caller passes row_ptr + k0 + kb.
__device__ __forceinline__ v16h ld_afrag(const _Float16* p) {
  v8h lo = *(const v8h*)p;
  v8h hi = *(const v8h*)(p + 16);
  return frag_cat(lo, hi);
}
// B-fragment (32x16 f16) from Bt[N][K]: lane holds col = lane&15, kb2 = (lane>>4)*8;
// data at [k0+kb2, +8) and [k0+16+kb2, +8). Caller passes col_ptr + k0 + kb2.
__device__ __forceinline__ v16h ld_bfrag(const _Float16* p) {
  v8h lo = *(const v8h*)p;
  v8h hi = *(const v8h*)(p + 16);
  return frag_cat(lo, hi);
}

// window-ordered token index -> original flat index (fuses roll + partition).
__device__ __forceinline__ int win_to_orig(int g) {
  int w = g >> 8, p = g & 255;
  int wd = w >> 6, wh = (w >> 3) & 7, ww = w & 7;
  int pz = p >> 6, py = (p >> 3) & 7, px = p & 7;
  int z = ((wd << 2) + pz + 2) & 15;
  int y = ((wh << 3) + py + 4) & 63;
  int x = ((ww << 3) + px + 4) & 63;
  return (z << 12) | (y << 6) | x;
}

// ---------------------------------------------------------------------------
// 1) fp32 [K][N] -> f16 transposed Bt [N][K]
// ---------------------------------------------------------------------------
__global__ void k_prep(const float* __restrict__ src, _Float16* __restrict__ dst,
                       int K, int N) {
  long t = (long)blockIdx.x * blockDim.x + threadIdx.x;
  if (t >= (long)K * N) return;
  int n = (int)(t % N), k = (int)(t / N);
  dst[(long)n * K + k] = (_Float16)src[(long)k * N + n];
}

// ---------------------------------------------------------------------------
// 2) mod[6*512] = silu(t_emb) @ ada_w + ada_b
// ---------------------------------------------------------------------------
__global__ void k_mod(const float* __restrict__ t_emb, const float* __restrict__ ada_w,
                      const float* __restrict__ ada_b, float* __restrict__ mod) {
  int j = blockIdx.x * blockDim.x + threadIdx.x;
  if (j >= 6 * DIMC) return;
  float acc = ada_b[j];
  for (int k = 0; k < DIMC; ++k) {
    float s = t_emb[k];
    acc += (s / (1.0f + __expf(-s))) * ada_w[(long)k * (6 * DIMC) + j];
  }
  mod[j] = acc;
}

// ---------------------------------------------------------------------------
// 3) bias[h][i][j] = rel_table[REL_IDX(i,j)][h], REL_IDX rebuilt on device
// ---------------------------------------------------------------------------
__global__ void k_bias(const float* __restrict__ table, float* __restrict__ bias) {
  int t = blockIdx.x * blockDim.x + threadIdx.x;  // 65536 = 256*256 pairs
  int i = t >> 8, j = t & 255;
  int dz = (i >> 6)        - (j >> 6)        + 3;   // 2*WD-1 = 7 values
  int dy = ((i >> 3) & 7)  - ((j >> 3) & 7)  + 7;   // 2*WH-1 = 15
  int dx = (i & 7)         - (j & 7)         + 7;   // 2*WW-1 = 15
  int idx = (dz * 15 + dy) * 15 + dx;
  #pragma unroll
  for (int h = 0; h < NHEAD; ++h)
    bias[(long)h * 65536 + t] = table[idx * NHEAD + h];
}

// ---------------------------------------------------------------------------
// 4) fused (optional roll+partition gather) + LayerNorm + AdaLN modulate -> f16
// ---------------------------------------------------------------------------
template<bool WINDOWED>
__global__ __launch_bounds__(256)
void k_ln(const float* __restrict__ xin, _Float16* __restrict__ xo,
          const float* __restrict__ shift, const float* __restrict__ scale) {
  int lane = threadIdx.x & 31, wv = threadIdx.x >> 5;
  int g = blockIdx.x * 8 + wv;
  int src = WINDOWED ? win_to_orig(g) : g;
  const float* xr = xin + (long)src * DIMC;
  float4 vbuf[4];
  float s = 0.f, sq = 0.f;
  #pragma unroll
  for (int i = 0; i < 4; ++i) {
    float4 v = *(const float4*)(xr + i * 128 + lane * 4);
    vbuf[i] = v;
    s  += v.x + v.y + v.z + v.w;
    sq += v.x * v.x + v.y * v.y + v.z * v.z + v.w * v.w;
  }
  #pragma unroll
  for (int off = 16; off >= 1; off >>= 1) {
    s  += __shfl_xor(s,  off, 32);
    sq += __shfl_xor(sq, off, 32);
  }
  float mean = s * (1.0f / DIMC);
  float var  = sq * (1.0f / DIMC) - mean * mean;
  float rstd = rsqrtf(var + 1e-5f);
  _Float16* orow = xo + (long)g * DIMC;
  #pragma unroll
  for (int i = 0; i < 4; ++i) {
    int c = i * 128 + lane * 4;
    float4 v = vbuf[i];
    v4h o;
    o[0] = (_Float16)((v.x - mean) * rstd * (1.0f + scale[c + 0]) + shift[c + 0]);
    o[1] = (_Float16)((v.y - mean) * rstd * (1.0f + scale[c + 1]) + shift[c + 1]);
    o[2] = (_Float16)((v.z - mean) * rstd * (1.0f + scale[c + 2]) + shift[c + 2]);
    o[3] = (_Float16)((v.w - mean) * rstd * (1.0f + scale[c + 3]) + shift[c + 3]);
    *(v4h*)(orow + c) = o;
  }
}

// ---------------------------------------------------------------------------
// 5) WMMA GEMM: C[M][N] = A[M][K](f16, row-major) x Bt[N][K](f16) + bias.
//    Block = 256 thr = 8 waves (4x2), block tile 128x64, wave tile 32x32.
//    MODE 0: store f16 (+bias)                              (QKV)
//    MODE 1: store f16, exact GELU                          (FC1)
//    MODE 2: f32 out[win_to_orig(row)] = resid + gate*(v)   (proj + reverse)
//    MODE 3: f32 out[row]              = resid + gate*(v)   (FC2 + residual)
// ---------------------------------------------------------------------------
template<int MODE>
__global__ __launch_bounds__(256)
void k_gemm(const _Float16* __restrict__ A, const _Float16* __restrict__ Bt,
            const float* __restrict__ bias,
            _Float16* __restrict__ outh, float* __restrict__ outf,
            const float* __restrict__ resid, const float* __restrict__ gate,
            int M, int N, int K) {
  const int lane = threadIdx.x & 31;
  const int wv   = threadIdx.x >> 5;
  const int l16  = lane & 15;
  const int lh   = lane >> 4;
  const long r0  = (long)blockIdx.x * 128 + (wv & 3) * 32;
  const long n0  = (long)blockIdx.y * 64  + (wv >> 2) * 32;
  const int akb  = lh * 8;
  const int bkb  = lh * 8;
  const _Float16* a0p = A  + (r0 + l16) * K + akb;
  const _Float16* a1p = a0p + (long)16 * K;
  const _Float16* b0p = Bt + (n0 + l16) * K + bkb;
  const _Float16* b1p = b0p + (long)16 * K;
  v8f c00 = {}, c01 = {}, c10 = {}, c11 = {};
  for (int k0 = 0; k0 < K; k0 += 32) {
    v16h a0 = ld_afrag(a0p + k0);
    v16h a1 = ld_afrag(a1p + k0);
    v16h b0 = ld_bfrag(b0p + k0);
    v16h b1 = ld_bfrag(b1p + k0);
    c00 = WMMA_F16(a0, b0, c00);
    c01 = WMMA_F16(a0, b1, c01);
    c10 = WMMA_F16(a1, b0, c10);
    c11 = WMMA_F16(a1, b1, c11);
  }
  v8f accs[4] = {c00, c01, c10, c11};
  #pragma unroll
  for (int t = 0; t < 4; ++t) {
    const long rb = r0 + (t >> 1) * 16 + lh * 8;
    const long c  = n0 + (t & 1) * 16 + l16;
    const float bv = bias[c];
    #pragma unroll
    for (int j = 0; j < 8; ++j) {
      const long row = rb + j;
      float v = accs[t][j] + bv;
      if (MODE == 0) {
        outh[row * N + c] = (_Float16)v;
      } else if (MODE == 1) {
        float ge = 0.5f * v * (1.0f + erff(v * 0.70710678118654752f));
        outh[row * N + c] = (_Float16)ge;
      } else if (MODE == 2) {
        int dst = win_to_orig((int)row);
        long idx = (long)dst * DIMC + c;
        outf[idx] = resid[idx] + gate[c] * v;
      } else {
        long idx = row * DIMC + c;
        outf[idx] = resid[idx] + gate[c] * v;
      }
    }
  }
}

// ---------------------------------------------------------------------------
// 6) Windowed attention, one block per (window, head). 256 thr = 8 waves,
//    each wave owns 32 query rows. Vt staged in LDS. Softmax restructured so
//    cross-lane reductions happen once per row-tile (not per key-tile):
//      pass1: per-lane elementwise max across all key tiles, one xor-reduce
//      pass2: p = exp(s-m) (unnormalized) -> P@V WMMAs; per-lane partial row
//             sums accumulate for free; O is scaled by 1/rowsum at the end.
// ---------------------------------------------------------------------------
#define VT_STRIDE 264   // 256 + 8 pad (f16), 16B-aligned row stride
#define P_STRIDE  40    // 32 + 8 pad (f16)

__global__ __launch_bounds__(256)
void k_attn(const _Float16* __restrict__ qkv, const float* __restrict__ biasbuf,
            _Float16* __restrict__ out) {
  __shared__ __align__(16) _Float16 vt[64 * VT_STRIDE];
  __shared__ __align__(16) _Float16 ptile[8 * 16 * P_STRIDE];

  const int w = blockIdx.x >> 3;
  const int h = blockIdx.x & 7;
  const int tid = threadIdx.x, lane = tid & 31, wv = tid >> 5;
  const int l16 = lane & 15, lh = lane >> 4;
  const int akb = lh * 8, bkb = lh * 8;
  const _Float16* base = qkv + (long)w * NWIN * (3 * DIMC) + h * HDIM;

  // stage V^T into LDS: vt[c][p] = V[p][c]
  for (int e = tid; e < NWIN * HDIM; e += 256) {
    int p = e >> 6, c = e & 63;
    vt[c * VT_STRIDE + p] = base[(long)p * (3 * DIMC) + 2 * DIMC + c];
  }
  __syncthreads();

  // Q fragments for this wave's 32 rows (2 row-tiles x 2 K-steps)
  const int r0 = wv * 32;
  v16h qf[2][2];
  #pragma unroll
  for (int rt = 0; rt < 2; ++rt)
    #pragma unroll
    for (int s = 0; s < 2; ++s)
      qf[rt][s] = ld_afrag(base + (long)(r0 + rt * 16 + l16) * (3 * DIMC) + s * 32 + akb);

  const float* biasrow = biasbuf + (long)h * 65536;
  const float scale = 0.125f;  // 1/sqrt(64)

  // ---- pass 1: per-lane partial row max over all 16 key tiles ----
  float mst[2][8];
  #pragma unroll
  for (int rt = 0; rt < 2; ++rt)
    #pragma unroll
    for (int j = 0; j < 8; ++j) mst[rt][j] = -1e30f;

  for (int rt = 0; rt < 2; ++rt) {
    const int rr = r0 + rt * 16 + lh * 8;
    for (int nt = 0; nt < 16; ++nt) {
      v8f sacc = {};
      #pragma unroll
      for (int s = 0; s < 2; ++s) {
        v16h kf = ld_bfrag(base + (long)(nt * 16 + l16) * (3 * DIMC) + DIMC + s * 32 + bkb);
        sacc = WMMA_F16(qf[rt][s], kf, sacc);
      }
      const int n = nt * 16 + l16;
      #pragma unroll
      for (int j = 0; j < 8; ++j)
        mst[rt][j] = fmaxf(mst[rt][j], sacc[j] * scale + biasrow[(rr + j) * 256 + n]);
    }
  }
  // single cross-lane max reduction (xor over the 16-lane half)
  #pragma unroll
  for (int rt = 0; rt < 2; ++rt)
    #pragma unroll
    for (int j = 0; j < 8; ++j)
      #pragma unroll
      for (int off = 8; off >= 1; off >>= 1)
        mst[rt][j] = fmaxf(mst[rt][j], __shfl_xor(mst[rt][j], off, 32));

  // ---- pass 2: unnormalized P = exp(S-m), O += P @ V, partial row sums ----
  _Float16* pw = ptile + wv * 16 * P_STRIDE;
  v8f oacc[2][4];
  float lpart[2][8];
  #pragma unroll
  for (int rt = 0; rt < 2; ++rt) {
    #pragma unroll
    for (int ct = 0; ct < 4; ++ct) oacc[rt][ct] = (v8f){};
    #pragma unroll
    for (int j = 0; j < 8; ++j) lpart[rt][j] = 0.f;
  }

  for (int rt = 0; rt < 2; ++rt) {
    const int rr = r0 + rt * 16 + lh * 8;
    for (int nt2 = 0; nt2 < 8; ++nt2) {
      #pragma unroll
      for (int half = 0; half < 2; ++half) {
        const int nt = nt2 * 2 + half;
        v8f sacc = {};
        #pragma unroll
        for (int s = 0; s < 2; ++s) {
          v16h kf = ld_bfrag(base + (long)(nt * 16 + l16) * (3 * DIMC) + DIMC + s * 32 + bkb);
          sacc = WMMA_F16(qf[rt][s], kf, sacc);
        }
        const int n = nt * 16 + l16;
        #pragma unroll
        for (int j = 0; j < 8; ++j) {
          float sv = sacc[j] * scale + biasrow[(rr + j) * 256 + n];
          float p = __expf(sv - mst[rt][j]);
          lpart[rt][j] += p;
          pw[(lh * 8 + j) * P_STRIDE + half * 16 + l16] = (_Float16)p;
        }
      }
      // wave-private LDS tile: drain DS pipe before re-reading in A-layout
      asm volatile("s_wait_dscnt 0" ::: "memory");
      v8h plo = *(const v8h*)(pw + l16 * P_STRIDE + akb);
      v8h phi = *(const v8h*)(pw + l16 * P_STRIDE + 16 + akb);
      v16h pf = frag_cat(plo, phi);
      #pragma unroll
      for (int ct = 0; ct < 4; ++ct) {
        const _Float16* vp = vt + (ct * 16 + l16) * VT_STRIDE + nt2 * 32 + bkb;
        v16h vf = frag_cat(*(const v8h*)vp, *(const v8h*)(vp + 16));
        oacc[rt][ct] = WMMA_F16(pf, vf, oacc[rt][ct]);
      }
    }
  }

  // ---- deferred normalization: one sum reduction per row-tile ----
  float linv[2][8];
  #pragma unroll
  for (int rt = 0; rt < 2; ++rt)
    #pragma unroll
    for (int j = 0; j < 8; ++j) {
      float l = lpart[rt][j];
      #pragma unroll
      for (int off = 8; off >= 1; off >>= 1) l += __shfl_xor(l, off, 32);
      linv[rt][j] = 1.0f / l;
    }

  // ---- store O = (P_unnorm @ V) / rowsum (f16, window order) ----
  #pragma unroll
  for (int rt = 0; rt < 2; ++rt)
    #pragma unroll
    for (int ct = 0; ct < 4; ++ct)
      #pragma unroll
      for (int j = 0; j < 8; ++j) {
        int row = r0 + rt * 16 + lh * 8 + j;
        out[(long)(w * NWIN + row) * DIMC + h * HDIM + ct * 16 + l16] =
            (_Float16)(oacc[rt][ct][j] * linv[rt][j]);
      }
}

// ---------------------------------------------------------------------------
// launch
// ---------------------------------------------------------------------------
extern "C" void kernel_launch(void* const* d_in, const int* in_sizes, int n_in,
                              void* d_out, int out_size, void* d_ws, size_t ws_size,
                              hipStream_t stream) {
  const float* x      = (const float*)d_in[0];
  const float* t_emb  = (const float*)d_in[1];
  const float* ada_w  = (const float*)d_in[2];
  const float* ada_b  = (const float*)d_in[3];
  const float* rtab   = (const float*)d_in[4];
  const float* qkv_w  = (const float*)d_in[5];
  const float* qkv_b  = (const float*)d_in[6];
  const float* proj_w = (const float*)d_in[7];
  const float* proj_b = (const float*)d_in[8];
  const float* fc1_w  = (const float*)d_in[9];
  const float* fc1_b  = (const float*)d_in[10];
  const float* fc2_w  = (const float*)d_in[11];
  const float* fc2_b  = (const float*)d_in[12];
  float* out = (float*)d_out;

  char* ws = (char*)d_ws;
  size_t off = 0;
  auto take = [&](size_t bytes) -> char* {
    char* p = ws + off;
    off = (off + bytes + 255) & ~(size_t)255;
    return p;
  };
  _Float16* qkv_wt  = (_Float16*)take((size_t)1536 * 512 * 2);
  _Float16* proj_wt = (_Float16*)take((size_t)512 * 512 * 2);
  _Float16* fc1_wt  = (_Float16*)take((size_t)2048 * 512 * 2);
  _Float16* fc2_wt  = (_Float16*)take((size_t)512 * 2048 * 2);
  float*    mod     = (float*)take((size_t)6 * DIMC * 4);
  float*    biasbuf = (float*)take((size_t)NHEAD * 65536 * 4);
  _Float16* xn      = (_Float16*)take((size_t)LTOK * DIMC * 2);       // xn1, later xn2
  _Float16* qkvbuf  = (_Float16*)take((size_t)LTOK * 3 * DIMC * 2);   // later: h1 (low)
  _Float16* attnout = (_Float16*)take((size_t)LTOK * DIMC * 2);       // later: h1 (high)
  float*    xres    = (float*)take((size_t)LTOK * DIMC * 4);
  _Float16* h1      = qkvbuf;  // 192MB + 64MB regions recycled = exactly 256MB

  // mod split: shift_msa, scale_msa, gate_msa, shift_mlp, scale_mlp, gate_mlp
  const float* shift_msa = mod + 0 * DIMC;
  const float* scale_msa = mod + 1 * DIMC;
  const float* gate_msa  = mod + 2 * DIMC;
  const float* shift_mlp = mod + 3 * DIMC;
  const float* scale_mlp = mod + 4 * DIMC;
  const float* gate_mlp  = mod + 5 * DIMC;

  // 1) f16 transposed weights
  k_prep<<<(512 * 1536 + 255) / 256, 256, 0, stream>>>(qkv_w, qkv_wt, 512, 1536);
  k_prep<<<(512 * 512 + 255) / 256, 256, 0, stream>>>(proj_w, proj_wt, 512, 512);
  k_prep<<<(512 * 2048 + 255) / 256, 256, 0, stream>>>(fc1_w, fc1_wt, 512, 2048);
  k_prep<<<(2048 * 512 + 255) / 256, 256, 0, stream>>>(fc2_w, fc2_wt, 2048, 512);
  // 2) AdaLN modulation vector
  k_mod<<<(6 * DIMC + 255) / 256, 256, 0, stream>>>(t_emb, ada_w, ada_b, mod);
  // 3) relative position bias
  k_bias<<<65536 / 256, 256, 0, stream>>>(rtab, biasbuf);
  // 4) roll + partition + LN + modulate (MSA)
  k_ln<true><<<LTOK / 8, 256, 0, stream>>>(x, xn, shift_msa, scale_msa);
  // 5) QKV GEMM -> f16 [LTOK][1536] (window order)
  k_gemm<0><<<dim3(LTOK / 128, 1536 / 64), 256, 0, stream>>>(
      xn, qkv_wt, qkv_b, qkvbuf, nullptr, nullptr, nullptr, LTOK, 1536, 512);
  // 6) windowed attention (256 windows x 8 heads)
  k_attn<<<256 * NHEAD, 256, 0, stream>>>(qkvbuf, biasbuf, attnout);
  // 7) proj GEMM + window-reverse/roll-back scatter + gated residual -> xres f32
  k_gemm<2><<<dim3(LTOK / 128, 512 / 64), 256, 0, stream>>>(
      attnout, proj_wt, proj_b, nullptr, xres, x, gate_msa, LTOK, 512, 512);
  // 8) LN + modulate (MLP)
  k_ln<false><<<LTOK / 8, 256, 0, stream>>>(xres, xn, shift_mlp, scale_mlp);
  // 9) FC1 GEMM + exact GELU -> f16 hidden [LTOK][2048]
  k_gemm<1><<<dim3(LTOK / 128, 2048 / 64), 256, 0, stream>>>(
      xn, fc1_wt, fc1_b, h1, nullptr, nullptr, nullptr, LTOK, 2048, 512);
  // 10) FC2 GEMM + gated residual -> d_out f32
  k_gemm<3><<<dim3(LTOK / 128, 512 / 64), 256, 0, stream>>>(
      h1, fc2_wt, fc2_b, nullptr, out, xres, gate_mlp, LTOK, 512, 2048);
}
